// OMStackBot_74758200754683
// MI455X (gfx1250) — matrix operand order repaired and checked by reading, compiled, not run on hardware
//
#include <hip/hip_runtime.h>
#include <hip/hip_bf16.h>

typedef __attribute__((ext_vector_type(16))) _Float16 v16h;
typedef __attribute__((ext_vector_type(8)))  _Float16 v8h;
typedef __attribute__((ext_vector_type(8)))  float    v8f;

#define DD 512
#define TT 64

// ---------------------------------------------------------------------------
// fp32 -> f16 convert (grid-stride)
// ---------------------------------------------------------------------------
__global__ __launch_bounds__(256)
void cvt_f16(const float* __restrict__ s, _Float16* __restrict__ d, long n)
{
    for (long i = (long)blockIdx.x * 256 + threadIdx.x; i < n;
         i += (long)gridDim.x * 256)
        d[i] = (_Float16)s[i];
}

// ---------------------------------------------------------------------------
// GEMM: C[M,N] = act(A[M,K] @ W[K,N] + bias). A,W f16, accumulate fp32 via
// v_wmma_f32_16x16x32_f16. Output fp32 (Cf) or f16 (Ch).
// Block = 256 threads (8 waves), block tile 64(M) x 64(N).
// Wave w: M-tile = w&3, N-tiles 2*(w>>2) and 2*(w>>2)+1.
// act: 0=none, 1=relu, 2=leaky(0.3) (branchless negative-slope select).
// Inner loop uses pointer increments (A += 32 halves, B += 32*N halves) so
// addressing is a single 64-bit add per stream instead of a per-iter multiply.
// ---------------------------------------------------------------------------
__global__ __launch_bounds__(256)
void gemm_h(const _Float16* __restrict__ A, const _Float16* __restrict__ W,
            const float* __restrict__ bias,
            float* __restrict__ Cf, _Float16* __restrict__ Ch,
            int M, int N, int K, int act)
{
    const int lane = threadIdx.x & 31;
    const int wave = threadIdx.x >> 5;
    const int grp  = lane >> 4;
    const int l15  = lane & 15;

    const int m0 = blockIdx.y * 64 + (wave & 3) * 16;
    const int nb = blockIdx.x * 64 + (wave >> 2) * 32;

    v8f acc0 = {};
    v8f acc1 = {};

    // A 16x32 f16 tile: lane row = m0+l15; halves 0..7 at K=k0+grp*8,
    // halves 8..15 at K=k0+16+grp*8 (ISA 16-bit A layout).
    const _Float16* ap = A + (size_t)(m0 + l15) * K + grp * 8;
    // B 32x16 tile: lane holds column N=nb+l15; halves 0..7 at K=k0+grp*8,
    // halves 8..15 at K=k0+16+grp*8 (same k order as A).
    const _Float16* bp = W + (size_t)(grp * 8) * N + nb + l15;
    const size_t bstep = (size_t)32 * N;

#pragma unroll 2
    for (int k0 = 0; k0 < K; k0 += 32) {
        v8h alo = *reinterpret_cast<const v8h*>(ap);
        v8h ahi = *reinterpret_cast<const v8h*>(ap + 16);
        v16h av = __builtin_shufflevector(alo, ahi,
            0,1,2,3,4,5,6,7,8,9,10,11,12,13,14,15);

        v16h bv0, bv1;
#pragma unroll
        for (int s = 0; s < 16; ++s) {
            const size_t o = (size_t)(s < 8 ? s : s + 8) * N;
            bv0[s] = bp[o];
            bv1[s] = bp[o + 16];
        }

        acc0 = __builtin_amdgcn_wmma_f32_16x16x32_f16(false, av, false, bv0,
                                                      (short)0, acc0, false, false);
        acc1 = __builtin_amdgcn_wmma_f32_16x16x32_f16(false, av, false, bv1,
                                                      (short)0, acc1, false, false);
        ap += 32;
        bp += bstep;
    }

    const int col0 = nb + l15;
    const int col1 = nb + 16 + l15;
    const float bb0 = bias[col0];
    const float bb1 = bias[col1];
    const float nslope = (act == 0) ? 1.f : ((act == 1) ? 0.f : 0.3f);
    const int rbase = m0 + grp * 8;
#pragma unroll
    for (int r = 0; r < 8; ++r) {
        float v0 = acc0[r] + bb0;
        float v1 = acc1[r] + bb1;
        v0 = (v0 > 0.f) ? v0 : v0 * nslope;
        v1 = (v1 > 0.f) ? v1 : v1 * nslope;
        const size_t i0 = (size_t)(rbase + r) * N + col0;
        const size_t i1 = (size_t)(rbase + r) * N + col1;
        if (Cf) { Cf[i0] = v0; Cf[i1] = v1; }
        else    { Ch[i0] = (_Float16)v0; Ch[i1] = (_Float16)v1; }
    }
}

// ---------------------------------------------------------------------------
// Helpers
// ---------------------------------------------------------------------------
__device__ __forceinline__ float block_sum256(float v, float* sred)
{
    for (int o = 16; o > 0; o >>= 1) v += __shfl_down(v, o, 32);
    const int w = threadIdx.x >> 5;
    if ((threadIdx.x & 31) == 0) sred[w] = v;
    __syncthreads();
    if (threadIdx.x == 0) {
        float r = sred[0];
        for (int i = 1; i < 8; ++i) r += sred[i];
        sred[8] = r;
    }
    __syncthreads();
    float r = sred[8];
    __syncthreads();
    return r;
}

__device__ __forceinline__ float sigm(float x) { return 1.f / (1.f + __expf(-x)); }

__global__ __launch_bounds__(256)
void emb_gather(const int* __restrict__ ids, const float* __restrict__ tab,
                float* __restrict__ te)
{
    const int b = blockIdx.x;
    const int id = ids[b];
    for (int j = threadIdx.x; j < DD; j += 256)
        te[(size_t)b * DD + j] = tab[(size_t)id * DD + j];
}

__global__ __launch_bounds__(256)
void copy_f32(const float* __restrict__ s, float* __restrict__ d, int n)
{
    int i = blockIdx.x * 256 + threadIdx.x;
    if (i < n) d[i] = s[i];
}

// dst(f16)[row] = [s0 | s1 | s2 | s3] (fp32 sources, per-row strides)
__global__ __launch_bounds__(256)
void concat4_h(_Float16* __restrict__ dst, int dstride,
               const float* __restrict__ s0, int st0, int n0,
               const float* __restrict__ s1, int st1, int n1,
               const float* __restrict__ s2, int st2, int n2,
               const float* __restrict__ s3, int st3, int n3)
{
    const int b = blockIdx.x;
    _Float16* d = dst + (size_t)b * dstride;
    for (int j = threadIdx.x; j < n0; j += 256) d[j]         = (_Float16)s0[(size_t)b*st0 + j];
    for (int j = threadIdx.x; j < n1; j += 256) d[n0+j]      = (_Float16)s1[(size_t)b*st1 + j];
    for (int j = threadIdx.x; j < n2; j += 256) d[n0+n1+j]   = (_Float16)s2[(size_t)b*st2 + j];
    if (s3)
        for (int j = threadIdx.x; j < n3; j += 256)
            d[n0+n1+n2+j] = (_Float16)s3[(size_t)b*st3 + j];
}

// dist_in row r=(b*3+s): [inp_enc(b) | te(b) | m(b,s) | cand(b,s)] -> f16
__global__ __launch_bounds__(256)
void concat_dist_h(_Float16* __restrict__ dst, const float* __restrict__ oinp,
                   const float* __restrict__ te, const float* __restrict__ m,
                   const float* __restrict__ cand, int t)
{
    const int r = blockIdx.x, b = r / 3, s = r % 3;
    _Float16* d = dst + (size_t)r * 2048;
    const float* ie = oinp + ((size_t)b * TT + t) * DD;
    const float* tb = te + (size_t)b * DD;
    const float* mm = m + (size_t)b * 1536 + s * DD;
    const float* cc = cand + (size_t)b * 1536 + s * DD;
    for (int j = threadIdx.x; j < DD; j += 256) {
        d[j]        = (_Float16)ie[j];
        d[DD + j]   = (_Float16)tb[j];
        d[1024 + j] = (_Float16)mm[j];
        d[1536 + j] = (_Float16)cc[j];
    }
}

// com cell tail: gates = sigmoid(LN(a2[:, :1536]; g,beta)); h = LN_na(vg*v+hg*h+cg*cell)
__global__ __launch_bounds__(256)
void com_gate(const float* __restrict__ a2,
              const float* __restrict__ v, int vstride,
              const float* __restrict__ hi, int hstride,
              const float* __restrict__ g, const float* __restrict__ beta,
              float* __restrict__ out, int ostride)
{
    __shared__ float sred[9];
    __shared__ float s[DD];
    const int b = blockIdx.x;
    const float* arow = a2 + (size_t)b * 2048;

    float ls = 0.f, lq = 0.f;
    for (int j = threadIdx.x; j < 1536; j += 256) { float x = arow[j]; ls += x; lq += x*x; }
    const float mu  = block_sum256(ls, sred) * (1.f/1536.f);
    const float ex2 = block_sum256(lq, sred) * (1.f/1536.f);
    const float rs  = rsqrtf(ex2 - mu*mu + 0.001f);

    const float* vr = v  + (size_t)b * vstride;
    const float* hr = hi + (size_t)b * hstride;
    for (int j = threadIdx.x; j < DD; j += 256) {
        float vg = sigm((arow[j]        - mu)*rs*g[j]        + beta[j]);
        float hg = sigm((arow[DD  + j]  - mu)*rs*g[DD  + j]  + beta[DD  + j]);
        float cg = sigm((arow[1024 + j] - mu)*rs*g[1024 + j] + beta[1024 + j]);
        s[j] = vg*vr[j] + hg*hr[j] + cg*arow[1536 + j];
    }
    __syncthreads();
    float s1 = 0.f, s2 = 0.f;
    for (int j = threadIdx.x; j < DD; j += 256) { float x = s[j]; s1 += x; s2 += x*x; }
    const float mu2 = block_sum256(s1, sred) * (1.f/512.f);
    const float v2  = block_sum256(s2, sred) * (1.f/512.f) - mu2*mu2;
    const float rs2 = rsqrtf(v2 + 0.001f);
    float* orow = out + (size_t)b * ostride;
    for (int j = threadIdx.x; j < DD; j += 256) orow[j] = (s[j] - mu2) * rs2;
}

// dec cell tail
__global__ __launch_bounds__(256)
void dec_gate(const float* __restrict__ a2,
              const float* __restrict__ parent, int pstride,
              const float* __restrict__ g, const float* __restrict__ beta,
              float* __restrict__ out, int ostride)
{
    __shared__ float sred[9];
    __shared__ float s[DD];
    const int b = blockIdx.x;
    const float* arow = a2 + (size_t)b * 1536;

    float ls = 0.f, lq = 0.f;
    for (int j = threadIdx.x; j < 1024; j += 256) { float x = arow[j]; ls += x; lq += x*x; }
    const float mu  = block_sum256(ls, sred) * (1.f/1024.f);
    const float ex2 = block_sum256(lq, sred) * (1.f/1024.f);
    const float rs  = rsqrtf(ex2 - mu*mu + 0.001f);

    const float* pr = parent + (size_t)b * pstride;
    for (int j = threadIdx.x; j < DD; j += 256) {
        float gt = sigm((arow[j]       - mu)*rs*g[j]      + beta[j]);
        float cg = sigm((arow[DD + j]  - mu)*rs*g[DD + j] + beta[DD + j]);
        s[j] = gt*pr[j] + cg*arow[1024 + j];
    }
    __syncthreads();
    float s1 = 0.f, s2 = 0.f;
    for (int j = threadIdx.x; j < DD; j += 256) { float x = s[j]; s1 += x; s2 += x*x; }
    const float mu2 = block_sum256(s1, sred) * (1.f/512.f);
    const float v2  = block_sum256(s2, sred) * (1.f/512.f) - mu2*mu2;
    const float rs2 = rsqrtf(v2 + 0.001f);
    float* orow = out + (size_t)b * ostride;
    for (int j = threadIdx.x; j < DD; j += 256) orow[j] = (s[j] - mu2) * rs2;
}

// beta -> stick-breaking p_hat/p/rcp, memory blend, chl[:,0] = m[:,0]
__global__ __launch_bounds__(256)
void dist_p(const float* __restrict__ h1, const float* __restrict__ Wd2,
            const float* __restrict__ bd2,
            float* __restrict__ m, const float* __restrict__ cand,
            float* __restrict__ p, float* __restrict__ phats,
            float* __restrict__ chl, int t)
{
    __shared__ float bet[3], srcp[3];
    const int b = blockIdx.x, tid = threadIdx.x, w = tid >> 5, lane = tid & 31;
    if (w < 3) {
        const float* row = h1 + (size_t)(b*3 + w) * DD;
        float acc = 0.f;
        for (int j = lane; j < DD; j += 32) acc += row[j] * Wd2[j];
        for (int o = 16; o > 0; o >>= 1) acc += __shfl_down(acc, o, 32);
        if (lane == 0) bet[w] = sigm(acc + bd2[0]);
    }
    __syncthreads();
    if (tid == 0) {
        const float b0 = bet[0], b1 = bet[1], b2 = bet[2];
        const float ph0 = (1.f-b2)*(1.f-b1)*(1.f-b0);
        const float ph1 = b0*(1.f-b2)*(1.f-b1);
        const float ph2 = b1*(1.f-b2);
        const float ph3 = b2;
        const float inv = 1.f / (ph1 + ph2 + ph3);
        const float p0 = ph1*inv, p1 = ph2*inv, p2 = ph3*inv;
        p[b*3+0] = p0; p[b*3+1] = p1; p[b*3+2] = p2;
        srcp[0] = p0 + p1 + p2; srcp[1] = p1 + p2; srcp[2] = p2;
        float* ph = phats + ((size_t)b * TT + t) * 4;
        ph[0] = ph0; ph[1] = ph1; ph[2] = ph2; ph[3] = ph3;
    }
    __syncthreads();
    float* mrow = m + (size_t)b * 1536;
    const float* crow = cand + (size_t)b * 1536;
    for (int j = tid; j < 1536; j += 256) {
        const float r = srcp[j >> 9];
        const float nm = mrow[j] * (1.f - r) + crow[j] * r;
        mrow[j] = nm;
        if (j < DD) chl[(size_t)b * 1536 + j] = nm;
    }
}

// output head: out3 @ Wa + softmax + done-slot insertion + log
__global__ __launch_bounds__(256)
void out_head(const float* __restrict__ p, const float* __restrict__ chl,
              const float* __restrict__ te, const float* __restrict__ oinp,
              const float* __restrict__ Wa, const float* __restrict__ ba,
              const float* __restrict__ phats, float* __restrict__ logits, int t)
{
    __shared__ float o3[1536];
    __shared__ float lg[16];
    const int b = blockIdx.x, tid = threadIdx.x, w = tid >> 5, lane = tid & 31;
    const float* ch = chl + (size_t)b * 1536;
    const float p0 = p[b*3], p1 = p[b*3+1], p2 = p[b*3+2];
    for (int j = tid; j < DD; j += 256)
        o3[j] = p0*ch[j] + p1*ch[DD + j] + p2*ch[1024 + j];
    for (int j = tid; j < DD; j += 256) o3[DD + j] = te[(size_t)b * DD + j];
    const float* ie = oinp + ((size_t)b * TT + t) * DD;
    for (int j = tid; j < DD; j += 256) o3[1024 + j] = ie[j];
    __syncthreads();
    for (int c = w; c < 15; c += 8) {
        float acc = 0.f;
        for (int j = lane; j < 1536; j += 32) acc += o3[j] * Wa[(size_t)j * 15 + c];
        for (int o = 16; o > 0; o >>= 1) acc += __shfl_down(acc, o, 32);
        if (lane == 0) lg[c] = acc + ba[c];
    }
    __syncthreads();
    if (tid == 0) {
        float mx = lg[0];
        for (int c = 1; c < 15; ++c) mx = fmaxf(mx, lg[c]);
        float e[15]; float sum = 0.f;
        for (int c = 0; c < 15; ++c) { e[c] = __expf(lg[c] - mx); sum += e[c]; }
        float pe = phats[((size_t)b * TT + t) * 4];
        pe = fminf(fmaxf(pe, 1e-6f), 1.f - 1e-6f);
        const float sc = (1.f - pe) / sum;
        float* L = logits + ((size_t)b * TT + t) * 16;
        for (int c = 0; c < 5; ++c)  L[c]     = __logf(e[c] * sc);
        L[5] = __logf(pe);
        for (int c = 5; c < 15; ++c) L[c + 1] = __logf(e[c] * sc);
    }
}

// ---------------------------------------------------------------------------
extern "C" void kernel_launch(void* const* d_in, const int* in_sizes, int n_in,
                              void* d_out, int out_size, void* d_ws, size_t ws_size,
                              hipStream_t stream)
{
    const float* obs      = (const float*)d_in[0];
    const int*   env_ids  = (const int*)  d_in[1];
    const float* prev_m   = (const float*)d_in[2];
    const float* emb_tab  = (const float*)d_in[4];
    const float* W_obs    = (const float*)d_in[5];
    const float* b_obs    = (const float*)d_in[6];
    const float* com_W1   = (const float*)d_in[7];
    const float* com_b1   = (const float*)d_in[8];
    const float* com_W2   = (const float*)d_in[9];
    const float* com_b2   = (const float*)d_in[10];
    const float* com_g    = (const float*)d_in[11];
    const float* com_beta = (const float*)d_in[12];
    const float* dec_W1   = (const float*)d_in[13];
    const float* dec_b1   = (const float*)d_in[14];
    const float* dec_W2   = (const float*)d_in[15];
    const float* dec_b2   = (const float*)d_in[16];
    const float* dec_g    = (const float*)d_in[17];
    const float* dec_beta = (const float*)d_in[18];
    const float* Wd1      = (const float*)d_in[19];
    const float* bd1      = (const float*)d_in[20];
    const float* Wd2      = (const float*)d_in[21];
    const float* bd2      = (const float*)d_in[22];
    const float* Wa       = (const float*)d_in[23];
    const float* ba       = (const float*)d_in[24];

    // ---- fp32 workspace ----
    float* ws   = (float*)d_ws;
    float* te   = ws;  ws += 64 * 512;
    float* oinp = ws;  ws += 4096 * 512;
    float* m    = ws;  ws += 64 * 1536;
    float* a2   = ws;  ws += 64 * 2048;
    float* cand = ws;  ws += 64 * 1536;
    float* h1   = ws;  ws += 192 * 512;
    float* chl  = ws;  ws += 64 * 1536;
    float* a2d  = ws;  ws += 64 * 1536;
    float* pbuf = ws;  ws += 64 * 4;

    // ---- f16 workspace (weights converted once per launch + activations) ----
    _Float16* hw = (_Float16*)ws;
    _Float16* obsh    = hw;  hw += (size_t)4096 * 128;
    _Float16* wobsh   = hw;  hw += (size_t)128 * 512;
    _Float16* comW1h  = hw;  hw += (size_t)3 * 2048 * 2048;
    _Float16* comW2h  = hw;  hw += (size_t)3 * 2048 * 2048;
    _Float16* decW1h  = hw;  hw += (size_t)2 * 1536 * 2048;
    _Float16* decW2h  = hw;  hw += (size_t)2 * 2048 * 1536;
    _Float16* wd1h    = hw;  hw += (size_t)2048 * 512;
    _Float16* xcomh   = hw;  hw += (size_t)64 * 2048;
    _Float16* a1h     = hw;  hw += (size_t)64 * 2048;
    _Float16* dinh    = hw;  hw += (size_t)192 * 2048;
    _Float16* xdech   = hw;  hw += (size_t)64 * 1536;
    _Float16* a1dh    = hw;  hw += (size_t)64 * 2048;

    float* logits = (float*)d_out;             // [B][T][16]
    float* phats  = (float*)d_out + 64*64*16;  // [B][T][4]

    // one-time (per launch) f32 -> f16 conversions; weights reused 64 steps
    cvt_f16<<<2048, 256, 0, stream>>>(obs,    obsh,   (long)4096 * 128);
    cvt_f16<<<128,  256, 0, stream>>>(W_obs,  wobsh,  (long)128 * 512);
    cvt_f16<<<4096, 256, 0, stream>>>(com_W1, comW1h, (long)3 * 2048 * 2048);
    cvt_f16<<<4096, 256, 0, stream>>>(com_W2, comW2h, (long)3 * 2048 * 2048);
    cvt_f16<<<4096, 256, 0, stream>>>(dec_W1, decW1h, (long)2 * 1536 * 2048);
    cvt_f16<<<4096, 256, 0, stream>>>(dec_W2, decW2h, (long)2 * 2048 * 1536);
    cvt_f16<<<1024, 256, 0, stream>>>(Wd1,    wd1h,   (long)2048 * 512);

    emb_gather<<<64, 256, 0, stream>>>(env_ids, emb_tab, te);
    copy_f32<<<(64*1536 + 255)/256, 256, 0, stream>>>(prev_m, m, 64*1536);
    // obs_inp = obs @ W_obs + b_obs : (4096 x 128) @ (128 x 512)
    gemm_h<<<dim3(512/64, 4096/64), 256, 0, stream>>>(obsh, wobsh, b_obs,
        oinp, (_Float16*)nullptr, 4096, 512, 128, 0);

    for (int t = 0; t < TT; ++t) {
        // ---- composition cells, i = 2, 1, 0 ----
        for (int i = 2; i >= 0; --i) {
            const float* vptr; int vstr;
            if (i == 2) { vptr = oinp + (size_t)t * DD; vstr = TT * DD; }
            else        { vptr = cand + (size_t)(i + 1) * DD; vstr = 1536; }
            concat4_h<<<64, 256, 0, stream>>>(xcomh, 2048,
                vptr, vstr, DD,
                m + i * DD, 1536, DD,
                oinp + (size_t)t * DD, TT * DD, DD,
                te, DD, DD);
            gemm_h<<<dim3(32, 1), 256, 0, stream>>>(xcomh,
                comW1h + (size_t)i * 2048 * 2048, com_b1 + i * 2048,
                (float*)nullptr, a1h, 64, 2048, 2048, 1);
            gemm_h<<<dim3(32, 1), 256, 0, stream>>>(a1h,
                comW2h + (size_t)i * 2048 * 2048, com_b2 + i * 2048,
                a2, (_Float16*)nullptr, 64, 2048, 2048, 0);
            com_gate<<<64, 256, 0, stream>>>(a2, vptr, vstr, m + i * DD, 1536,
                com_g + i * 1536, com_beta + i * 1536, cand + i * DD, 1536);
        }
        // ---- distribution head + memory blend ----
        concat_dist_h<<<192, 256, 0, stream>>>(dinh, oinp, te, m, cand, t);
        gemm_h<<<dim3(512/64, 192/64), 256, 0, stream>>>(dinh, wd1h, bd1,
            h1, (_Float16*)nullptr, 192, 512, 2048, 2);
        dist_p<<<64, 256, 0, stream>>>(h1, Wd2, bd2, m, cand, pbuf, phats, chl, t);
        // ---- decomposition cells, i = 0, 1 ----
        for (int i = 0; i < 2; ++i) {
            concat4_h<<<64, 256, 0, stream>>>(xdech, 1536,
                chl + i * DD, 1536, DD,
                oinp + (size_t)t * DD, TT * DD, DD,
                te, DD, DD,
                (const float*)nullptr, 0, 0);
            gemm_h<<<dim3(32, 1), 256, 0, stream>>>(xdech,
                decW1h + (size_t)i * 1536 * 2048, dec_b1 + i * 2048,
                (float*)nullptr, a1dh, 64, 2048, 1536, 1);
            gemm_h<<<dim3(24, 1), 256, 0, stream>>>(a1dh,
                decW2h + (size_t)i * 2048 * 1536, dec_b2 + i * 1536,
                a2d, (_Float16*)nullptr, 64, 1536, 2048, 0);
            dec_gate<<<64, 256, 0, stream>>>(a2d, chl + i * DD, 1536,
                dec_g + i * 1024, dec_beta + i * 1024, chl + (i + 1) * DD, 1536);
        }
        // ---- action head ----
        out_head<<<64, 256, 0, stream>>>(pbuf, chl, te, oinp, Wa, ba, phats, logits, t);
    }
    (void)in_sizes; (void)n_in; (void)out_size; (void)ws_size;
}
